// Memory_22376779612420
// MI455X (gfx1250) — compile-verified, matrix operand order checked by reading, not run on hardware
//
#include <hip/hip_runtime.h>
#include <math.h>

// ---------------------------------------------------------------------------
// Problem constants (from the reference): B=4, Ck=128, Cv=512, T=8, H=W=32
// ---------------------------------------------------------------------------
#define HW    1024      // H*W
#define THW   8192      // T*H*W
#define CK    128
#define CV    512
#define NB    4

typedef __attribute__((ext_vector_type(16))) __bf16        v16bf;
typedef __attribute__((ext_vector_type(8)))  float         v8f;
typedef __attribute__((ext_vector_type(8)))  unsigned int  v8ui;

// fp32 -> bf16 with round-to-nearest-even
__device__ __forceinline__ unsigned int f2bf(float f) {
    unsigned int u = __builtin_bit_cast(unsigned int, f);
    u = u + 0x7FFFu + ((u >> 16) & 1u);
    return u >> 16;
}
__device__ __forceinline__ unsigned int pk2(float a, float b) {
    return f2bf(a) | (f2bf(b) << 16);
}
// K offset of VGPR v for 16-bit A/B WMMA fragments (ISA 7.12.2), hi = lane>=16
__device__ __forceinline__ int kbase(int v, int hi) {
    return ((v & 4) << 2) | ((v & 3) << 1) | (hi << 3);
}
__device__ __forceinline__ float sigmoidf_(float x) {
    return 1.0f / (1.0f + __expf(-x));
}

// Load the 8 b128 runs of a 16x128 bf16 A-operand row (4 K-chunks of 32).
__device__ __forceinline__ void load_frag8(uint4* f, const unsigned short* row, int hi) {
    #pragma unroll
    for (int kc = 0; kc < 4; ++kc) {
        f[2 * kc]     = *(const uint4*)(row + kc * 32 + hi * 8);       // K 0..7
        f[2 * kc + 1] = *(const uint4*)(row + kc * 32 + 16 + hi * 8);  // K 16..23
    }
}
__device__ __forceinline__ v16bf frag_ab(const uint4* f, int kc) {
    const uint4 a0 = f[2 * kc], a1 = f[2 * kc + 1];
    const v8ui u = {a0.x, a0.y, a0.z, a0.w, a1.x, a1.y, a1.z, a1.w};
    return __builtin_bit_cast(v16bf, u);
}

// ---------------------------------------------------------------------------
// Prepass A: values_m fp32 [b][cv][n] -> bf16 (same layout). 8 floats/thread.
// ---------------------------------------------------------------------------
__global__ __launch_bounds__(256) void vconv_kernel(const float* __restrict__ vm,
                                                    unsigned short* __restrict__ vb) {
    const size_t i = ((size_t)blockIdx.x * 256 + threadIdx.x) * 8;
    const float4 f0 = *(const float4*)(vm + i);
    const float4 f1 = *(const float4*)(vm + i + 4);
    uint4 o;
    o.x = pk2(f0.x, f0.y); o.y = pk2(f0.z, f0.w);
    o.z = pk2(f1.x, f1.y); o.w = pk2(f1.z, f1.w);
    *(uint4*)(vb + i) = o;
}

// ---------------------------------------------------------------------------
// Prepass B: keys_m fp32 [b][c][n] -> bf16 transposed [b][n][c] (LDS tiled).
// Grid: 4 b * 4 ctiles * 256 ntiles = 4096 blocks x 256 threads.
// ---------------------------------------------------------------------------
__global__ __launch_bounds__(256) void ktrans_kernel(const float* __restrict__ km,
                                                     unsigned short* __restrict__ kmT) {
    const int blk = blockIdx.x;
    const int b  = blk >> 10;
    const int c0 = ((blk >> 8) & 3) * 32;
    const int n0 = (blk & 255) * 32;
    __shared__ float t[32][33];
    const int tx = threadIdx.x & 31, ty = threadIdx.x >> 5;   // ty: 0..7
    const float* src = km + (size_t)b * CK * THW;
    #pragma unroll
    for (int i = 0; i < 4; ++i) {
        const int c = c0 + ty + i * 8;
        t[ty + i * 8][tx] = src[(size_t)c * THW + n0 + tx];
    }
    __syncthreads();
    const int nloc = threadIdx.x >> 3, d = threadIdx.x & 7;   // d: 4 c-values each
    unsigned int* dst =
        (unsigned int*)(kmT + ((size_t)b * THW + n0 + nloc) * CK + c0);
    dst[d * 2]     = pk2(t[d * 4 + 0][nloc], t[d * 4 + 1][nloc]);
    dst[d * 2 + 1] = pk2(t[d * 4 + 2][nloc], t[d * 4 + 3][nloc]);
}

// ---------------------------------------------------------------------------
// Kernel 1: 7x7 Gaussian blur (sigma=1.4, reflect-101) of the mask.
// ---------------------------------------------------------------------------
__global__ __launch_bounds__(256) void blur_kernel(const float* __restrict__ mask,
                                                   float* __restrict__ pblur) {
    const int b   = blockIdx.x;
    const int tid = threadIdx.x;
    __shared__ int flag;
    if (tid == 0) flag = 0;
    __syncthreads();
    int any = 0;
    for (int p = tid; p < HW; p += 256)
        if (mask[b * HW + p] != 0.0f) any = 1;
    if (any) atomicOr(&flag, 1);
    __syncthreads();
    const bool nz = (flag != 0);

    float g[7];
    float s = 0.f;
    #pragma unroll
    for (int i = 0; i < 7; ++i) {
        float d = (float)(i - 3);
        g[i] = __expf(-(d * d) / (2.0f * 1.4f * 1.4f));
        s += g[i];
    }
    #pragma unroll
    for (int i = 0; i < 7; ++i) g[i] /= s;

    for (int p = tid; p < HW; p += 256) {
        const int y = p >> 5, x = p & 31;
        float acc = 0.f;
        #pragma unroll
        for (int dy = 0; dy < 7; ++dy) {
            int yy = y + dy - 3;
            yy = yy < 0 ? -yy : (yy > 31 ? 62 - yy : yy);
            #pragma unroll
            for (int dx = 0; dx < 7; ++dx) {
                int xx = x + dx - 3;
                xx = xx < 0 ? -xx : (xx > 31 ? 62 - xx : xx);
                const float mv = nz ? mask[b * HW + yy * 32 + xx] : 1.0f;
                acc += g[dy] * g[dx] * mv;
            }
        }
        pblur[b * HW + p] = acc;
    }
}

// ---------------------------------------------------------------------------
// Kernel 2: two 3x3 convs over concat(key_q, pblur) [129 ch], sigmoid, gate.
// ---------------------------------------------------------------------------
__global__ __launch_bounds__(256) void conv_kernel(const float* __restrict__ keyq,
                                                   const float* __restrict__ pblur,
                                                   const float* __restrict__ cw,
                                                   const float* __restrict__ cwb,
                                                   const float* __restrict__ cb,
                                                   const float* __restrict__ cbb,
                                                   float* __restrict__ pmf) {
    const int idx = blockIdx.x * 256 + threadIdx.x;      // 0 .. 4095
    const int b = idx >> 10, p = idx & (HW - 1);
    const int y = p >> 5, x = p & 31;
    float aw = cwb[0], ab = cbb[0];
    for (int c = 0; c < CK + 1; ++c) {
        const float* chan = (c < CK) ? (keyq + ((size_t)b * CK + c) * HW)
                                     : (pblur + (size_t)b * HW);
        #pragma unroll
        for (int dy = -1; dy <= 1; ++dy) {
            const int yy = y + dy;
            if (yy < 0 || yy > 31) continue;
            #pragma unroll
            for (int dx = -1; dx <= 1; ++dx) {
                const int xx = x + dx;
                if (xx < 0 || xx > 31) continue;
                const float val = chan[yy * 32 + xx];
                const int wi = c * 9 + (dy + 1) * 3 + (dx + 1);
                aw += cw[wi] * val;
                ab += cb[wi] * val;
            }
        }
    }
    pmf[b * HW + p] = sigmoidf_(aw) * pblur[b * HW + p] + sigmoidf_(ab);
}

// ---------------------------------------------------------------------------
// Kernel 3: copy value_q into output channels [512, 1024)
// ---------------------------------------------------------------------------
__global__ __launch_bounds__(256) void copy_vq_kernel(const float* __restrict__ vq,
                                                      float* __restrict__ out) {
    const size_t i = (size_t)blockIdx.x * 256 + threadIdx.x;   // 4*512*1024 total
    const size_t b = i / (CV * HW), r = i % (CV * HW);
    out[b * (2 * CV * HW) + (size_t)CV * HW + r] = vq[i];
}

// ---------------------------------------------------------------------------
// Kernel 4: fused flash-attention memory read (bf16 WMMA, f32 accumulate).
// A-operands pre-converted to bf16 in workspace (kmT transposed, vb row-
// major): fragments are pure global_load_b128. The t0/t1 value fragments are
// issued right after the S-GEMM so they are in flight across the softmax
// barriers; t2/t3 are double-buffered between WMMA groups.
// Grid: 256 blocks (b, 16-query tile) x 256 threads (8 waves).
// ---------------------------------------------------------------------------
__global__ __launch_bounds__(256) void attn_kernel(const unsigned short* __restrict__ kmT,
                                                   const unsigned short* __restrict__ vb,
                                                   const float* __restrict__ keyq,
                                                   const float* __restrict__ pmf,
                                                   float* __restrict__ out) {
    const int tid  = threadIdx.x;
    const int w    = tid >> 5;
    const int lane = tid & 31;
    const int lo   = lane & 15;
    const int hi   = lane >> 4;
    const int b    = blockIdx.x >> 6;
    const int q0   = (blockIdx.x & 63) << 4;

    const unsigned short* kmb = kmT + (size_t)b * THW * CK;   // [n][c] bf16
    const unsigned short* vbb = vb  + (size_t)b * CV * THW;   // [cv][n] bf16
    const float* kq = keyq + (size_t)b * CK * HW;             // [c][q] f32
    float*      op  = out  + (size_t)b * (2 * CV * HW);

    __shared__ __align__(16) unsigned short sP[16 * 128];  // P tile [q=16][n=128] bf16
    __shared__ float sMax[8][16];
    __shared__ float sSum[8][16];

    // --- kq B-fragments (pre-scaled by 1/sqrt(128)), resident in VGPRs ---
    v16bf bq[4];
    {
        const float scale = 0.08838834764831845f;
        #pragma unroll
        for (int kc = 0; kc < 4; ++kc) {
            v8ui u;
            #pragma unroll
            for (int v = 0; v < 8; ++v) {
                const int c = kc * 32 + kbase(v, hi);
                const float f0 = kq[(size_t)c * HW + q0 + lo] * scale;
                const float f1 = kq[(size_t)(c + 1) * HW + q0 + lo] * scale;
                u[v] = pk2(f0, f1);
            }
            bq[kc] = __builtin_bit_cast(v16bf, u);
        }
    }

    v8f acc0 = {}, acc1 = {}, acc2 = {}, acc3 = {};
    float run_max = -1e30f, l = 0.0f;
    const int wbase = w * 64;
    const unsigned short* vrow0 = vbb + (size_t)(wbase + lo) * THW;  // t=0 row

    for (int it = 0; it < 64; ++it) {
        const int n0 = it << 7;
        const unsigned short* vr = vrow0 + n0;

        // ---- S tile: issue all 8 km loads first, then the 4 WMMAs ----
        v8f S = {};
        {
            uint4 kf[8];
            const unsigned short* kr = kmb + (size_t)(n0 + w * 16 + lo) * CK;
            load_frag8(kf, kr, hi);
            #pragma unroll
            for (int kc = 0; kc < 4; ++kc)
                S = __builtin_amdgcn_wmma_f32_16x16x32_bf16(
                        false, frag_ab(kf, kc), false, bq[kc],
                        (short)0, S, false, false);
        }

        // ---- issue t0/t1 value fragments now: in flight across the softmax ----
        uint4 fa[8], fb[8];
        load_frag8(fa, vr, hi);                     // t0
        load_frag8(fb, vr + 16 * THW, hi);          // t1

        // ---- per-column (query) max within this wave's tile ----
        float cmax = S[0];
        #pragma unroll
        for (int r = 1; r < 8; ++r) cmax = fmaxf(cmax, S[r]);
        cmax = fmaxf(cmax, __shfl_xor(cmax, 16, 32));

        __syncthreads();                 // B0: prior iter's LDS reads done
        sMax[w][lo] = cmax;              // both half-lanes write same value
        __syncthreads();                 // B1

        float nmax = run_max;
        #pragma unroll
        for (int w2 = 0; w2 < 8; ++w2) nmax = fmaxf(nmax, sMax[w2][lo]);
        const float sf = __expf(run_max - nmax);
        run_max = nmax;
        l *= sf;
        #pragma unroll
        for (int r = 0; r < 8; ++r) {
            acc0[r] *= sf; acc1[r] *= sf; acc2[r] *= sf; acc3[r] *= sf;
        }

        // ---- P = exp(S - max), column sums, stage P tile to LDS (bf16) ----
        float csum = 0.f;
        float pv[8];
        #pragma unroll
        for (int r = 0; r < 8; ++r) { pv[r] = __expf(S[r] - nmax); csum += pv[r]; }
        csum += __shfl_xor(csum, 16, 32);
        sSum[w][lo] = csum;              // both half-lanes write same value
        #pragma unroll
        for (int r = 0; r < 8; r += 2)   // rows r,r+1 are adjacent n -> dword
            *(unsigned int*)&sP[lo * 128 + w * 16 + r + hi * 8] = pk2(pv[r], pv[r + 1]);
        __syncthreads();                 // B2

        float cs = 0.f;
        #pragma unroll
        for (int w2 = 0; w2 < 8; ++w2) cs += sSum[w2][lo];
        l += cs;

        // ---- B-fragments of P from LDS (one ds_load_b32 per VGPR) ----
        v16bf pb[4];
        #pragma unroll
        for (int kc = 0; kc < 4; ++kc) {
            v8ui u;
            #pragma unroll
            for (int v = 0; v < 8; ++v) {
                const int kk = kc * 32 + kbase(v, hi);
                u[v] = *(const unsigned int*)&sP[lo * 128 + kk];
            }
            pb[kc] = __builtin_bit_cast(v16bf, u);
        }

        // ---- value GEMM, software-pipelined over the 4 cv tiles ----
        #pragma unroll
        for (int kc = 0; kc < 4; ++kc)
            acc0 = __builtin_amdgcn_wmma_f32_16x16x32_bf16(
                    false, frag_ab(fa, kc), false, pb[kc], (short)0, acc0, false, false);
        load_frag8(fa, vr + 32 * THW, hi);          // t2
        #pragma unroll
        for (int kc = 0; kc < 4; ++kc)
            acc1 = __builtin_amdgcn_wmma_f32_16x16x32_bf16(
                    false, frag_ab(fb, kc), false, pb[kc], (short)0, acc1, false, false);
        load_frag8(fb, vr + 48 * THW, hi);          // t3
        #pragma unroll
        for (int kc = 0; kc < 4; ++kc)
            acc2 = __builtin_amdgcn_wmma_f32_16x16x32_bf16(
                    false, frag_ab(fa, kc), false, pb[kc], (short)0, acc2, false, false);
        #pragma unroll
        for (int kc = 0; kc < 4; ++kc)
            acc3 = __builtin_amdgcn_wmma_f32_16x16x32_bf16(
                    false, frag_ab(fb, kc), false, pb[kc], (short)0, acc3, false, false);

        if (it + 1 < 64) {   // gfx1250 global_prefetch_b8 of next chunk
            __builtin_prefetch(vrow0 + n0 + 128, 0, 1);
            __builtin_prefetch(kmb + (size_t)(n0 + 128 + w * 16 + lo) * CK, 0, 1);
        }
    }

    // ---- epilogue: softmax normalize, mask gate, store mem channels ----
    const float gate = pmf[(size_t)b * HW + q0 + lo] / l;
    #pragma unroll
    for (int r = 0; r < 8; ++r) {
        const int cv = wbase + r + hi * 8;
        op[(size_t)(cv)      * HW + q0 + lo] = acc0[r] * gate;
        op[(size_t)(cv + 16) * HW + q0 + lo] = acc1[r] * gate;
        op[(size_t)(cv + 32) * HW + q0 + lo] = acc2[r] * gate;
        op[(size_t)(cv + 48) * HW + q0 + lo] = acc3[r] * gate;
    }
}

// ---------------------------------------------------------------------------
extern "C" void kernel_launch(void* const* d_in, const int* in_sizes, int n_in,
                              void* d_out, int out_size, void* d_ws, size_t ws_size,
                              hipStream_t stream) {
    (void)in_sizes; (void)n_in; (void)out_size; (void)ws_size;
    const float* keys = (const float*)d_in[0];   // [4,128,8192]
    const float* vals = (const float*)d_in[1];   // [4,512,8192]
    const float* keyq = (const float*)d_in[2];   // [4,128,1024]
    const float* valq = (const float*)d_in[3];   // [4,512,1024]
    const float* mask = (const float*)d_in[4];   // [4,1024]
    const float* cw   = (const float*)d_in[5];   // [129*9]
    const float* cwb  = (const float*)d_in[6];   // [1]
    const float* cb   = (const float*)d_in[7];   // [129*9]
    const float* cbb  = (const float*)d_in[8];   // [1]
    float* out = (float*)d_out;                  // [4,1024,1024]

    // workspace layout
    float* pblur = (float*)d_ws;                           // 16 KB
    float* pmf   = pblur + NB * HW;                        // 16 KB
    unsigned short* kmT = (unsigned short*)(pmf + NB * HW);        // 8 MB  [b][n][c]
    unsigned short* vb  = kmT + (size_t)NB * THW * CK;             // 32 MB [b][cv][n]

    vconv_kernel<<<(NB * CV * THW) / (256 * 8), 256, 0, stream>>>(vals, vb);
    ktrans_kernel<<<NB * 4 * (THW / 32), 256, 0, stream>>>(keys, kmT);
    blur_kernel<<<NB, 256, 0, stream>>>(mask, pblur);
    conv_kernel<<<(NB * HW) / 256, 256, 0, stream>>>(keyq, pblur, cw, cwb, cb, cbb, pmf);
    copy_vq_kernel<<<(NB * CV * HW) / 256, 256, 0, stream>>>(valq, out);
    attn_kernel<<<NB * (HW / 16), 256, 0, stream>>>(kmT, vb, keyq, pmf, out);
}
